// CrossWinAttention_81673098101642
// MI455X (gfx1250) — compile-verified, matrix operand order checked by reading, not run on hardware
//
#include <hip/hip_runtime.h>

#define HEADS 4
#define DH 32
#define D 128
#define NROWS 256   // n*w1*w2 tokens per window
#define EPS 1e-5f

typedef __attribute__((ext_vector_type(16))) __bf16 v16bf;
typedef __attribute__((ext_vector_type(8)))  float  v8f;

union BFrag { v16bf v; unsigned short u[16]; };

// Native bf16 convert: let the backend emit v_cvt_*bf16_f32 instead of a
// hand-rolled RNE integer sequence.
__device__ __forceinline__ unsigned short f2bf(float f) {
  __bf16 h = (__bf16)f;
  return __builtin_bit_cast(unsigned short, h);
}
__device__ __forceinline__ float bf2f(unsigned short h) {
  return (float)__builtin_bit_cast(__bf16, h);
}
// 16-bit A/B fragment K index for element e (ISA 7.12.2): VGPR j=e/2 holds K pairs,
// lanes 16-31 are K+8; upper 4 VGPRs are K+16. Contiguous in e-runs -> LDS b128 loads.
__device__ __forceinline__ int kmap(int e, int half) {
  return ((e & 8) << 1) + (e & 6) + (half << 3) + (e & 1);
}

__device__ __forceinline__ float rmax16(float v) {
  v = fmaxf(v, __shfl_xor(v, 1, 32));
  v = fmaxf(v, __shfl_xor(v, 2, 32));
  v = fmaxf(v, __shfl_xor(v, 4, 32));
  v = fmaxf(v, __shfl_xor(v, 8, 32));
  return v;
}
__device__ __forceinline__ float rsum16(float v) {
  v += __shfl_xor(v, 1, 32);
  v += __shfl_xor(v, 2, 32);
  v += __shfl_xor(v, 4, 32);
  v += __shfl_xor(v, 8, 32);
  return v;
}

// LayerNorm 256 rows of 128 f32 -> bf16 LDS. One wave per row, 4 cols/lane.
__device__ __forceinline__ void ln_stage(const float* __restrict__ src,
                                         const float* __restrict__ g,
                                         const float* __restrict__ bta,
                                         unsigned short* dst,
                                         int bi, int win, int wv, int lane) {
  for (int row = wv; row < NROWS; row += 8) {
    int n = row >> 6, rw = row & 63;
    const float* p = src + ((((size_t)bi * 4 + n) * 64 + win) * 64 + rw) * D;
    float4 x = ((const float4*)p)[lane];
    float s  = x.x + x.y + x.z + x.w;
    float ss = x.x * x.x + x.y * x.y + x.z * x.z + x.w * x.w;
#pragma unroll
    for (int m = 1; m < 32; m <<= 1) {
      s  += __shfl_xor(s, m, 32);
      ss += __shfl_xor(ss, m, 32);
    }
    float mu   = s * (1.0f / 128.0f);
    float var  = ss * (1.0f / 128.0f) - mu * mu;
    float rstd = rsqrtf(var + EPS);
    int c = lane * 4;
    dst[row * D + c + 0] = f2bf((x.x - mu) * rstd * g[c + 0] + bta[c + 0]);
    dst[row * D + c + 1] = f2bf((x.y - mu) * rstd * g[c + 1] + bta[c + 1]);
    dst[row * D + c + 2] = f2bf((x.z - mu) * rstd * g[c + 2] + bta[c + 2]);
    dst[row * D + c + 3] = f2bf((x.w - mu) * rstd * g[c + 3] + bta[c + 3]);
  }
}

// Stage a 128x128 f32 weight matrix into LDS as bf16.
__device__ __forceinline__ void load_w(const float* __restrict__ W,
                                       unsigned short* sWs, int tid) {
  for (int i = tid; i < (D * D) / 4; i += 256) {
    float4 w = ((const float4*)W)[i];
    int o = i * 4;
    sWs[o + 0] = f2bf(w.x);
    sWs[o + 1] = f2bf(w.y);
    sWs[o + 2] = f2bf(w.z);
    sWs[o + 3] = f2bf(w.w);
  }
}

// [256x128] bf16 (LDS) @ [128x128] bf16 (LDS) + bias -> bf16 LDS, via WMMA.
__device__ __forceinline__ void gemm_qkv(const unsigned short* sAs,
                                         const unsigned short* sWs,
                                         const float* __restrict__ bias,
                                         unsigned short* dst,
                                         int wv, int lane) {
  int half = lane >> 4, l15 = lane & 15;
  for (int t = wv; t < 128; t += 8) {   // 16 m-tiles x 8 n-tiles
    int mt = t >> 3, nt = t & 7;
    v8f acc = {};
#pragma unroll
    for (int ks = 0; ks < 4; ++ks) {
      int kb = ks << 5;
      BFrag a, b;
#pragma unroll
      for (int e = 0; e < 16; ++e) {
        int k = kb + kmap(e, half);
        a.u[e] = sAs[(mt * 16 + l15) * D + k];
        b.u[e] = sWs[k * D + nt * 16 + l15];
      }
      acc = __builtin_amdgcn_wmma_f32_16x16x32_bf16(
          false, a.v, false, b.v, (short)0, acc, false, false);
    }
    float bv = bias[nt * 16 + l15];
#pragma unroll
    for (int j = 0; j < 8; ++j) {
      int row = mt * 16 + j + half * 8;   // D-tile layout: row = vgpr + 8*half
      dst[row * D + nt * 16 + l15] = f2bf(acc[j] + bv);
    }
  }
}

__global__ void __launch_bounds__(256)
cross_win_attn_kernel(const float* __restrict__ q, const float* __restrict__ k,
                      const float* __restrict__ v,
                      const float* __restrict__ ln_q_g, const float* __restrict__ ln_q_b,
                      const float* __restrict__ ln_k_g, const float* __restrict__ ln_k_b,
                      const float* __restrict__ ln_v_g, const float* __restrict__ ln_v_b,
                      const float* __restrict__ Wq, const float* __restrict__ bq,
                      const float* __restrict__ Wk, const float* __restrict__ bk,
                      const float* __restrict__ Wv, const float* __restrict__ bv,
                      const float* __restrict__ Wp, const float* __restrict__ bp,
                      float* __restrict__ out) {
  __shared__ unsigned short sA[NROWS * D];   // 64 KB: LN scratch / attn output
  __shared__ unsigned short sQ[NROWS * D];   // 64 KB
  __shared__ unsigned short sK[NROWS * D];   // 64 KB
  __shared__ unsigned short sV[NROWS * D];   // 64 KB
  __shared__ unsigned short sW[D * D];       // 32 KB staged weights
  __shared__ unsigned short sP[8][16 * 36];  // 9 KB per-wave P transpose scratch

  const int tid  = threadIdx.x;
  const int lane = tid & 31;
  const int wv_id = tid >> 5;
  const int half = lane >> 4;
  const int l15  = lane & 15;
  const int bi   = blockIdx.x >> 6;
  const int win  = blockIdx.x & 63;

  // ---- Q projection ----
  ln_stage(q, ln_q_g, ln_q_b, sA, bi, win, wv_id, lane);
  load_w(Wq, sW, tid);
  __syncthreads();
  gemm_qkv(sA, sW, bq, sQ, wv_id, lane);
  __syncthreads();

  // ---- K projection ----
  ln_stage(k, ln_k_g, ln_k_b, sA, bi, win, wv_id, lane);
  load_w(Wk, sW, tid);
  __syncthreads();
  gemm_qkv(sA, sW, bk, sK, wv_id, lane);
  __syncthreads();

  // ---- V projection ----
  ln_stage(v, ln_v_g, ln_v_b, sA, bi, win, wv_id, lane);
  load_w(Wv, sW, tid);
  __syncthreads();
  gemm_qkv(sA, sW, bv, sV, wv_id, lane);
  __syncthreads();

  // Stage Wp now (sW free; attention doesn't touch it).
  load_w(Wp, sW, tid);

  // ---- Attention: flash-style online softmax, one wave per (head, q-tile) ----
  const float scale = 0.17677669529663687f;  // 1/sqrt(32)
  unsigned short* pscr = &sP[wv_id][0];
  for (int t = wv_id; t < 64; t += 8) {
    int h  = t >> 4;       // head
    int mt = t & 15;       // q tile (16 rows)
    v8f acc0 = {}, acc1 = {};
    float m_i[8], l_i[8];
#pragma unroll
    for (int j = 0; j < 8; ++j) { m_i[j] = -1e30f; l_i[j] = 0.0f; }

    for (int kc = 0; kc < 8; ++kc) {       // 8 chunks of 32 keys
      int kbase = kc * 32;
      // S tiles: A = Q[mt*16.., h, :32] (K=dh=32), B = K^T per 16-key tile
      BFrag qa;
#pragma unroll
      for (int e = 0; e < 16; ++e)
        qa.u[e] = sQ[(mt * 16 + l15) * D + h * DH + kmap(e, half)];
      BFrag kb0, kb1;
#pragma unroll
      for (int e = 0; e < 16; ++e) {
        int kk = kmap(e, half);
        kb0.u[e] = sK[(kbase + l15) * D + h * DH + kk];
        kb1.u[e] = sK[(kbase + 16 + l15) * D + h * DH + kk];
      }
      v8f z = {};
      v8f s0 = __builtin_amdgcn_wmma_f32_16x16x32_bf16(
          false, qa.v, false, kb0.v, (short)0, z, false, false);
      v8f s1 = __builtin_amdgcn_wmma_f32_16x16x32_bf16(
          false, qa.v, false, kb1.v, (short)0, z, false, false);

#pragma unroll
      for (int j = 0; j < 8; ++j) {
        float a0 = s0[j] * scale;
        float a1 = s1[j] * scale;
        float cmax = rmax16(fmaxf(a0, a1));          // row max over 32 keys
        float mn   = fmaxf(m_i[j], cmax);
        float corr = __expf(m_i[j] - mn);
        m_i[j] = mn;
        float p0 = __expf(a0 - mn);
        float p1 = __expf(a1 - mn);
        l_i[j] = l_i[j] * corr + rsum16(p0 + p1);
        acc0[j] *= corr;
        acc1[j] *= corr;
        int prow = j + half * 8;                      // row inside the 16-row tile
        pscr[prow * 36 + l15]      = f2bf(p0);
        pscr[prow * 36 + 16 + l15] = f2bf(p1);
      }
      __builtin_amdgcn_wave_barrier();  // same-wave LDS ops are in-order; fence compiler

      // P @ V chunk: A = P[16x32], B = V[32 x 16] per dh half
      BFrag pa;
#pragma unroll
      for (int e = 0; e < 16; ++e)
        pa.u[e] = pscr[l15 * 36 + kmap(e, half)];
      BFrag vb0, vb1;
#pragma unroll
      for (int e = 0; e < 16; ++e) {
        int kk = kmap(e, half);
        vb0.u[e] = sV[(kbase + kk) * D + h * DH + l15];
        vb1.u[e] = sV[(kbase + kk) * D + h * DH + 16 + l15];
      }
      acc0 = __builtin_amdgcn_wmma_f32_16x16x32_bf16(
          false, pa.v, false, vb0.v, (short)0, acc0, false, false);
      acc1 = __builtin_amdgcn_wmma_f32_16x16x32_bf16(
          false, pa.v, false, vb1.v, (short)0, acc1, false, false);
    }

    // normalize and write attn output tile into sA (bf16)
#pragma unroll
    for (int j = 0; j < 8; ++j) {
      float inv = 1.0f / l_i[j];
      int row = mt * 16 + j + half * 8;
      sA[row * D + h * DH + l15]      = f2bf(acc0[j] * inv);
      sA[row * D + h * DH + 16 + l15] = f2bf(acc1[j] * inv);
    }
  }
  __syncthreads();

  // ---- Output projection with mean over n folded into A operand ----
  // z[rw, :] = (mean_n a[n*64+rw, :]) @ Wp + bp  -> 64 x 128 output per window
  for (int t = wv_id; t < 32; t += 8) {   // 4 m-tiles x 8 n-tiles
    int mt = t >> 3, nt = t & 7;
    v8f acc = {};
#pragma unroll
    for (int ks = 0; ks < 4; ++ks) {
      int kb = ks << 5;
      BFrag a, b;
#pragma unroll
      for (int e = 0; e < 16; ++e) {
        int kk = kb + kmap(e, half);
        int rw = mt * 16 + l15;
        float s = 0.25f * (bf2f(sA[(0 * 64 + rw) * D + kk]) +
                           bf2f(sA[(1 * 64 + rw) * D + kk]) +
                           bf2f(sA[(2 * 64 + rw) * D + kk]) +
                           bf2f(sA[(3 * 64 + rw) * D + kk]));
        a.u[e] = f2bf(s);
        b.u[e] = sW[kk * D + nt * 16 + l15];
      }
      acc = __builtin_amdgcn_wmma_f32_16x16x32_bf16(
          false, a.v, false, b.v, (short)0, acc, false, false);
    }
    int col = nt * 16 + l15;
    float bpv = bp[col];
#pragma unroll
    for (int j = 0; j < 8; ++j) {
      int rw = mt * 16 + j + half * 8;
      out[(((size_t)bi * 64 + win) * 64 + rw) * D + col] = acc[j] + bpv;
    }
  }
}

extern "C" void kernel_launch(void* const* d_in, const int* in_sizes, int n_in,
                              void* d_out, int out_size, void* d_ws, size_t ws_size,
                              hipStream_t stream) {
  (void)in_sizes; (void)n_in; (void)out_size; (void)d_ws; (void)ws_size;
  const float* q      = (const float*)d_in[0];
  const float* k      = (const float*)d_in[1];
  const float* v      = (const float*)d_in[2];
  const float* ln_q_g = (const float*)d_in[3];
  const float* ln_q_b = (const float*)d_in[4];
  const float* ln_k_g = (const float*)d_in[5];
  const float* ln_k_b = (const float*)d_in[6];
  const float* ln_v_g = (const float*)d_in[7];
  const float* ln_v_b = (const float*)d_in[8];
  const float* Wq = (const float*)d_in[9];
  const float* bq = (const float*)d_in[10];
  const float* Wk = (const float*)d_in[11];
  const float* bk = (const float*)d_in[12];
  const float* Wv = (const float*)d_in[13];
  const float* bv = (const float*)d_in[14];
  const float* Wp = (const float*)d_in[15];
  const float* bp = (const float*)d_in[16];

  dim3 grid(128);   // 2 batches * 64 windows
  dim3 block(256);  // 8 wave32
  cross_win_attn_kernel<<<grid, block, 0, stream>>>(
      q, k, v, ln_q_g, ln_q_b, ln_k_g, ln_k_b, ln_v_g, ln_v_b,
      Wq, bq, Wk, bk, Wv, bv, Wp, bp, (float*)d_out);
}
